// InteractionDecoder_30202210026028
// MI455X (gfx1250) — matrix operand order repaired and checked by reading, not compiled, not run on hardware
//
#include <hip/hip_runtime.h>
#include <hip/hip_bf16.h>

typedef __attribute__((ext_vector_type(16))) _Float16 v16h;
typedef __attribute__((ext_vector_type(8)))  _Float16 v8h;
typedef __attribute__((ext_vector_type(8)))  float    v8f;
typedef __attribute__((ext_vector_type(4)))  float    v4f;

#define LATENT 128
#define WAVES  8
#define TPB    (WAVES * 32)

union Frag { v16h h; unsigned u[8]; };

// A fragment (16x32 f16) of a row-major f16 matrix in LDS (row = M, contiguous K):
// halfs 0..15 = K kc*32 + 16*laneHi + [0..15].
static __device__ __forceinline__ v16h loadAfrag(const _Float16* row, int kc, int laneHi) {
  int c0 = kc * 32 + laneHi * 16;
  v8h lo = *(const v8h*)(row + c0);
  v8h hi = *(const v8h*)(row + c0 + 8);
  v16h r;
#pragma unroll
  for (int i = 0; i < 8; ++i) { r[i] = lo[i]; r[8 + i] = hi[i]; }
  return r;
}

// B fragment (32x16 f16) of X^T built from an f32 row of X in global memory:
// lane holds column e (= lane%16); halfs 0..15 = X[e][kc*32 + 16*laneHi + 0..15].
static __device__ __forceinline__ v16h loadBfragX(const float* xrow, int kc, int laneHi) {
  int c0 = kc * 32 + laneHi * 16;
  v4f f0 = *(const v4f*)(xrow + c0);
  v4f f1 = *(const v4f*)(xrow + c0 + 4);
  v4f f2 = *(const v4f*)(xrow + c0 + 8);
  v4f f3 = *(const v4f*)(xrow + c0 + 12);
  v16h b;
#pragma unroll
  for (int j = 0; j < 4; ++j) {
    b[j]      = (_Float16)f0[j];
    b[4 + j]  = (_Float16)f1[j];
    b[8 + j]  = (_Float16)f2[j];
    b[12 + j] = (_Float16)f3[j];
  }
  return b;
}

// Pack 8 f32 (bias+ReLU applied) into 4 dwords of f16 pairs.
static __device__ __forceinline__ void packH(const v8f& acc, const float* biasRow,
                                             unsigned* out4) {
  float h[8];
#pragma unroll
  for (int v = 0; v < 8; ++v) {
    float x = acc[v] + biasRow[v];
    h[v] = x > 0.0f ? x : 0.0f;
  }
#pragma unroll
  for (int j = 0; j < 4; ++j) {
    union { unsigned u; _Float16 f[2]; } p;
    p.f[0] = (_Float16)h[2 * j];
    p.f[1] = (_Float16)h[2 * j + 1];
    out4[j] = p.u;
  }
}

// Build layer-2 B fragment (H^T tile) for K-chunk kc from packed hidden pairs.
// hpk[mt][j] on lane (e, g) holds H[e][mt*16 + 8g + 2j .. +1] as an f16 pair.
static __device__ __forceinline__ v16h buildHfrag(const unsigned hpk[8][4], int kc,
                                                  int laneHi) {
  Frag b;
#pragma unroll
  for (int j = 0; j < 4; ++j) {
    unsigned ownA = hpk[2 * kc][j];
    unsigned ownB = hpk[2 * kc + 1][j];
    unsigned swA = __shfl_xor(ownA, 16);
    unsigned swB = __shfl_xor(ownB, 16);
    b.u[j]     = laneHi ? swB : ownA;  // rows 0..7  of tile 2kc+laneHi
    b.u[4 + j] = laneHi ? ownB : swA;  // rows 8..15 of tile 2kc+laneHi
  }
  return b.h;
}

// One MLP (128 -> 128 relu -> out<=16, transposed):
// returns C2^T accumulator: lane (e=lane%16, g) VGPR v holds out[e][v + 8g].
static __device__ __forceinline__ v8f mlpTile(const v16h bx[4],
                                              const _Float16* w1T,
                                              const float*    b1,
                                              const _Float16* w2T,
                                              int laneLo, int laneHi) {
  unsigned hpk[8][4];
#pragma unroll
  for (int mt = 0; mt < 8; ++mt) {
    v8f acc = {};
#pragma unroll
    for (int kc = 0; kc < 4; ++kc) {
      v16h a = loadAfrag(w1T + (mt * 16 + laneLo) * LATENT, kc, laneHi);
      acc = __builtin_amdgcn_wmma_f32_16x16x32_f16(false, a, false, bx[kc],
                                                   (short)0, acc, false, false);
    }
    packH(acc, b1 + mt * 16 + laneHi * 8, hpk[mt]);
  }
  v8f c2 = {};
#pragma unroll
  for (int kc = 0; kc < 4; ++kc) {
    v16h a = loadAfrag(w2T + laneLo * LATENT, kc, laneHi);
    v16h b = buildHfrag(hpk, kc, laneHi);
    c2 = __builtin_amdgcn_wmma_f32_16x16x32_f16(false, a, false, b,
                                                (short)0, c2, false, false);
  }
  return c2;
}

// ---------------- Node weight precompute: w[n] = MLP_nw(node_latent[n]) ----------
__global__ __launch_bounds__(TPB) void node_weight_kernel(
    const float* __restrict__ X, const float* __restrict__ W1,
    const float* __restrict__ b1, const float* __restrict__ W2,
    const float* __restrict__ b2, float* __restrict__ wnode, int N)
{
  __shared__ _Float16 sW1T[LATENT * LATENT];   // [hidden][k]
  __shared__ _Float16 sW2T[16 * LATENT];       // [out(pad16)][k]
  __shared__ float    sB1[LATENT];

  for (int i = threadIdx.x; i < LATENT * LATENT; i += TPB) {
    int n = i >> 7, k = i & 127;
    sW1T[i] = (_Float16)W1[k * LATENT + n];
  }
  for (int i = threadIdx.x; i < 16 * LATENT; i += TPB) {
    int n = i >> 7, k = i & 127;
    sW2T[i] = (n < 1) ? (_Float16)W2[k] : (_Float16)0.0f;
  }
  for (int i = threadIdx.x; i < LATENT; i += TPB) sB1[i] = b1[i];
  __syncthreads();

  const int wave = threadIdx.x >> 5;
  const int lane = threadIdx.x & 31;
  const int laneLo = lane & 15;
  const int laneHi = lane >> 4;
  const int nTiles = (N + 15) >> 4;
  const float bias2 = b2[0];

  for (int t = blockIdx.x * WAVES + wave; t < nTiles; t += gridDim.x * WAVES) {
    const int base = t << 4;
    int row = base + laneLo; if (row >= N) row = N - 1;
    const float* xr = X + (size_t)row * LATENT;
    v16h bx[4];
#pragma unroll
    for (int kc = 0; kc < 4; ++kc) bx[kc] = loadBfragX(xr, kc, laneHi);

    v8f c2 = mlpTile(bx, sW1T, sB1, sW2T, laneLo, laneHi);

    if (laneHi == 0) {                 // lane e holds out[e][0] in c2[0]
      int nIdx = base + laneLo;
      if (nIdx < N) wnode[nIdx] = c2[0] + bias2;
    }
  }
}

// ---------------- Fused edge kernel: 3 MLPs + vector math ------------------------
__global__ __launch_bounds__(TPB) void edge_kernel(
    const int*   __restrict__ edge_index,
    const float* __restrict__ senders_pos,
    const float* __restrict__ receivers_pos,
    const float* __restrict__ vecA,
    const float* __restrict__ vecB,
    const float* __restrict__ vecC,
    const float* __restrict__ X,
    const float* __restrict__ W1_0, const float* __restrict__ b1_0,
    const float* __restrict__ W2_0, const float* __restrict__ b2_0,
    const float* __restrict__ W1_1, const float* __restrict__ b1_1,
    const float* __restrict__ W2_1, const float* __restrict__ b2_1,
    const float* __restrict__ W1_2, const float* __restrict__ b1_2,
    const float* __restrict__ W2_2, const float* __restrict__ b2_2,
    const float* __restrict__ wnode,
    float* __restrict__ out, int E)
{
  __shared__ _Float16 sW1T[3][LATENT * LATENT];   // 96 KB
  __shared__ _Float16 sW2T[3][16 * LATENT];       // 12 KB
  __shared__ float    sB1[3][LATENT];             // 1.5 KB

  {
    const float* W1s[3] = {W1_0, W1_1, W1_2};
    const float* W2s[3] = {W2_0, W2_1, W2_2};
    const float* B1s[3] = {b1_0, b1_1, b1_2};
    const int    od[3]  = {3, 3, 1};
    for (int m = 0; m < 3; ++m) {
      for (int i = threadIdx.x; i < LATENT * LATENT; i += TPB) {
        int n = i >> 7, k = i & 127;
        sW1T[m][i] = (_Float16)W1s[m][k * LATENT + n];
      }
      for (int i = threadIdx.x; i < 16 * LATENT; i += TPB) {
        int n = i >> 7, k = i & 127;
        sW2T[m][i] = (n < od[m]) ? (_Float16)W2s[m][k * od[m] + n] : (_Float16)0.0f;
      }
      for (int i = threadIdx.x; i < LATENT; i += TPB) sB1[m][i] = B1s[m][i];
    }
  }
  __syncthreads();

  const int wave = threadIdx.x >> 5;
  const int lane = threadIdx.x & 31;
  const int laneLo = lane & 15;
  const int laneHi = lane >> 4;
  const int nTiles = (E + 15) >> 4;
  const int stride = gridDim.x * WAVES;

  const float bf0 = b2_0[0], bf1 = b2_0[1], bf2 = b2_0[2];
  const float ba0 = b2_1[0], ba1 = b2_1[1], ba2 = b2_1[2];
  const float bl0 = b2_2[0];

  for (int t = blockIdx.x * WAVES + wave; t < nTiles; t += stride) {
    const int base = t << 4;
    int row = base + laneLo; if (row >= E) row = E - 1;
    const float* xr = X + (size_t)row * LATENT;

    // X fragments: loaded once, reused by all 3 MLPs x 8 hidden tiles.
    v16h bx[4];
#pragma unroll
    for (int kc = 0; kc < 4; ++kc) bx[kc] = loadBfragX(xr, kc, laneHi);

    {   // prefetch next tile's latent rows
      int t2 = t + stride;
      if (t2 < nTiles) {
        int r2 = (t2 << 4) + laneLo; if (r2 >= E) r2 = E - 1;
        __builtin_prefetch(X + (size_t)r2 * LATENT + laneHi * 64, 0, 0);
      }
    }

    // Transposed MLPs: results land with lane e holding out[e][j] in VGPR j.
    v8f cF = mlpTile(bx, &sW1T[0][0], &sB1[0][0], &sW2T[0][0], laneLo, laneHi);
    v8f cA = mlpTile(bx, &sW1T[1][0], &sB1[1][0], &sW2T[1][0], laneLo, laneHi);
    v8f cL = mlpTile(bx, &sW1T[2][0], &sB1[2][0], &sW2T[2][0], laneLo, laneHi);

    if (laneHi == 0) {
      int e = base + laneLo;
      if (e < E) {
        float cf0 = cF[0] + bf0, cf1 = cF[1] + bf1, cf2 = cF[2] + bf2;
        float ca0 = cA[0] + ba0, ca1 = cA[1] + ba1, ca2 = cA[2] + ba2;
        float lam = cL[0] + bl0;

        int sIdx = edge_index[e];
        int rIdx = edge_index[E + e];
        float ws = wnode[sIdx];
        float wr = wnode[rIdx];

        size_t e3 = (size_t)e * 3;
        float va0 = vecA[e3], va1 = vecA[e3 + 1], va2 = vecA[e3 + 2];
        float vb0 = vecB[e3], vb1 = vecB[e3 + 1], vb2 = vecB[e3 + 2];
        float vc0 = vecC[e3], vc1 = vecC[e3 + 1], vc2 = vecC[e3 + 2];
        float sp0 = senders_pos[e3],   sp1 = senders_pos[e3 + 1],   sp2 = senders_pos[e3 + 2];
        float rp0 = receivers_pos[e3], rp1 = receivers_pos[e3 + 1], rp2 = receivers_pos[e3 + 2];

        float f0 = cf0 * va0 + cf1 * vb0 + cf2 * vc0;
        float f1 = cf0 * va1 + cf1 * vb1 + cf2 * vc1;
        float f2 = cf0 * va2 + cf1 * vb2 + cf2 * vc2;
        float a0 = ca0 * va0 + ca1 * vb0 + ca2 * vc0;
        float a1 = ca0 * va1 + ca1 * vb1 + ca2 * vc1;
        float a2 = ca0 * va2 + ca1 * vb2 + ca2 * vc2;

        float inv = 1.0f / (ws + wr);
        float l0 = rp0 - (ws * sp0 + wr * rp0) * inv;
        float l1 = rp1 - (ws * sp1 + wr * rp1) * inv;
        float l2 = rp2 - (ws * sp2 + wr * rp2) * inv;
        float g0 = f0 * lam, g1 = f1 * lam, g2 = f2 * lam;
        float t0 = l1 * g2 - l2 * g1;
        float t1 = l2 * g0 - l0 * g2;
        float t2 = l0 * g1 - l1 * g0;

        out[e3]     = f0;
        out[e3 + 1] = f1;
        out[e3 + 2] = f2;
        size_t off = (size_t)E * 3 + e3;
        out[off]     = a0 - t0;
        out[off + 1] = a1 - t1;
        out[off + 2] = a2 - t2;
      }
    }
  }
}

extern "C" void kernel_launch(void* const* d_in, const int* in_sizes, int n_in,
                              void* d_out, int out_size, void* d_ws, size_t ws_size,
                              hipStream_t stream) {
  const int*   edge_index    = (const int*)d_in[0];
  const float* senders_pos   = (const float*)d_in[1];
  const float* receivers_pos = (const float*)d_in[2];
  const float* vector_a      = (const float*)d_in[3];
  const float* vector_b      = (const float*)d_in[4];
  const float* vector_c      = (const float*)d_in[5];
  const float* inter_latent  = (const float*)d_in[6];
  const float* node_latent   = (const float*)d_in[7];
  const float* i1_W1 = (const float*)d_in[8],  *i1_b1 = (const float*)d_in[9];
  const float* i1_W2 = (const float*)d_in[10], *i1_b2 = (const float*)d_in[11];
  const float* i2_W1 = (const float*)d_in[12], *i2_b1 = (const float*)d_in[13];
  const float* i2_W2 = (const float*)d_in[14], *i2_b2 = (const float*)d_in[15];
  const float* fs_W1 = (const float*)d_in[16], *fs_b1 = (const float*)d_in[17];
  const float* fs_W2 = (const float*)d_in[18], *fs_b2 = (const float*)d_in[19];
  const float* nw_W1 = (const float*)d_in[20], *nw_b1 = (const float*)d_in[21];
  const float* nw_W2 = (const float*)d_in[22], *nw_b2 = (const float*)d_in[23];

  const int E = in_sizes[1] / 3;
  const int N = in_sizes[7] / LATENT;
  float* wnode = (float*)d_ws;

  // Pass 1: per-node weight (pure function of node latent: 50K evals, not 3.2M).
  int nodeTiles  = (N + 15) / 16;
  int nodeBlocks = (nodeTiles + WAVES - 1) / WAVES;
  if (nodeBlocks > 512) nodeBlocks = 512;
  node_weight_kernel<<<nodeBlocks, TPB, 0, stream>>>(
      node_latent, nw_W1, nw_b1, nw_W2, nw_b2, wnode, N);

  // Pass 2: fused edge decoder.
  int edgeTiles  = (E + 15) / 16;
  int edgeBlocks = (edgeTiles + WAVES - 1) / WAVES;
  if (edgeBlocks > 2048) edgeBlocks = 2048;
  edge_kernel<<<edgeBlocks, TPB, 0, stream>>>(
      edge_index, senders_pos, receivers_pos, vector_a, vector_b, vector_c,
      inter_latent,
      i1_W1, i1_b1, i1_W2, i1_b2,
      i2_W1, i2_b1, i2_W2, i2_b2,
      fs_W1, fs_b1, fs_W2, fs_b2,
      wnode, (float*)d_out, E);
}
